// MaskedSelfAttention_1382979469700
// MI455X (gfx1250) — matrix operand order repaired and checked by reading, not compiled, not run on hardware
//
#include <hip/hip_runtime.h>
#include <stdint.h>

typedef __attribute__((ext_vector_type(16))) __bf16 v16bf;
typedef __attribute__((ext_vector_type(8)))  float  v8f;
typedef __attribute__((ext_vector_type(4)))  unsigned int u32x4;
typedef unsigned short u16;

#define Bdim 4
#define Sdim 2048
#define Ddim 1024

__device__ __forceinline__ u16 f2bf(float f) {
  unsigned u = __float_as_uint(f);
  u += 0x7FFFu + ((u >> 16) & 1u);   // round-to-nearest-even
  return (u16)(u >> 16);
}

union FragBF { v16bf v; u32x4 q[2]; };

// ---------------------------------------------------------------------------
// bf16 WMMA GEMM, async-LDS staged, double buffered, last iteration peeled.
//   C[M x N] = A[M x Kd] * B^T(+bias), per-z batch.
//   A : bf16 row-major [M][Kd], lda.
//   B : bf16 stored [N][Kd] (K-contiguous per output column), row stride ldb.
//   C : bf16 (OUT_BF16) or fp32; OUT_TRANS stores C transposed per batch as
//       [B][N][S] (used to produce V^T for the attn*V GEMM).
// Block: 256 threads = 8 waves; tile BM=128 x BN=128, BK=32.
// Wave grid 4(M) x 2(N); each wave: 2x4 tiles of v_wmma_f32_16x16x32_bf16.
// Staging: global_load_async_to_lds_b128 (ASYNCcnt), 4 issues/thread/tile.
// ---------------------------------------------------------------------------
template<bool OUT_BF16, bool ADD_BIAS, bool OUT_TRANS>
__global__ __launch_bounds__(256) void gemm_wmma_async(
    const u16* __restrict__ Ag, const u16* __restrict__ Bg,
    const float* __restrict__ bias, void* __restrict__ Cout,
    int Kd, int lda, int ldb, int ldc,
    long long strideA, long long strideB, long long strideC, float scale)
{
  constexpr int BM = 128, BN = 128, BK = 32, LS = 40;   // LS: padded stride (u16)
  constexpr unsigned ABUF = BM * LS * 2;                 // bytes per A buffer
  constexpr unsigned BBUF = BN * LS * 2;                 // bytes per B buffer
  __shared__ __align__(16) u16 As[2][BM][LS];
  __shared__ __align__(16) u16 Bs[2][BN][LS];

  Ag += (size_t)blockIdx.z * strideA;
  Bg += (size_t)blockIdx.z * strideB;

  const int bm = blockIdx.y * BM;
  const int bn = blockIdx.x * BN;
  const int tid  = threadIdx.x;
  const int wave = tid >> 5, lane = tid & 31;
  const int wm = wave >> 1, wn = wave & 1;
  const int g = lane >> 4, l15 = lane & 15;

  // ---- async staging maps: 512 chunks of 16B per tile, 2 per thread ----
  const unsigned asbase = (unsigned)(uintptr_t)&As[0][0][0]; // LDS byte offset
  const unsigned bsbase = (unsigned)(uintptr_t)&Bs[0][0][0];
  unsigned voffA[2], ldsA[2], voffB[2], ldsB[2];
#pragma unroll
  for (int i = 0; i < 2; ++i) {
    int id = tid + i * 256;
    int r = id >> 2, c = id & 3;                       // r: 0..127, c: 0..3
    voffA[i] = (unsigned)((bm + r) * lda + c * 8);     // element offset (k0 added later)
    ldsA[i]  = asbase + (unsigned)(r * (LS * 2) + c * 16);
    voffB[i] = (unsigned)((bn + r) * ldb + c * 8);
    ldsB[i]  = bsbase + (unsigned)(r * (LS * 2) + c * 16);
  }

  auto issue = [&](int k0, int buf) {
    const unsigned ab = buf ? ABUF : 0u, bb = buf ? BBUF : 0u;
#pragma unroll
    for (int i = 0; i < 2; ++i) {
      unsigned va = (voffA[i] + (unsigned)k0) * 2u;
      unsigned la = ldsA[i] + ab;
      asm volatile("global_load_async_to_lds_b128 %0, %1, %2"
                   :: "v"(la), "v"(va), "s"(Ag) : "memory");
    }
#pragma unroll
    for (int i = 0; i < 2; ++i) {
      unsigned vb = (voffB[i] + (unsigned)k0) * 2u;
      unsigned lb = ldsB[i] + bb;
      asm volatile("global_load_async_to_lds_b128 %0, %1, %2"
                   :: "v"(lb), "v"(vb), "s"(Bg) : "memory");
    }
  };

  v8f acc[2][4];
#pragma unroll
  for (int i = 0; i < 2; ++i)
#pragma unroll
    for (int j = 0; j < 4; ++j) {
      v8f z = {0.f, 0.f, 0.f, 0.f, 0.f, 0.f, 0.f, 0.f};
      acc[i][j] = z;
    }

  // fragment loads per ISA 16-bit A/B VGPR layouts (b128 each) + 8 WMMAs
  auto compute = [&](int cur) {
    FragBF a[2], b[4];
#pragma unroll
    for (int mi = 0; mi < 2; ++mi) {
      int row = wm * 32 + mi * 16 + l15;
      a[mi].q[0] = *(const u32x4*)&As[cur][row][8 * g];       // K = 8g..8g+7
      a[mi].q[1] = *(const u32x4*)&As[cur][row][16 + 8 * g];  // K = 16+8g..
    }
#pragma unroll
    for (int ni = 0; ni < 4; ++ni) {
      int col = wn * 64 + ni * 16 + l15;
      b[ni].q[0] = *(const u32x4*)&Bs[cur][col][8 * g];       // K = 8g..8g+7
      b[ni].q[1] = *(const u32x4*)&Bs[cur][col][16 + 8 * g];
    }
#pragma unroll
    for (int mi = 0; mi < 2; ++mi)
#pragma unroll
      for (int ni = 0; ni < 4; ++ni)
        acc[mi][ni] = __builtin_amdgcn_wmma_f32_16x16x32_bf16(
            false, a[mi].v, false, b[ni].v, (short)0, acc[mi][ni], false, false);
  };

  issue(0, 0);                                          // prologue
  int k0 = 0;
  for (; k0 < Kd - BK; k0 += BK) {                      // steady state (no branch)
    const int cur = (k0 >> 5) & 1;
    issue(k0 + BK, cur ^ 1);                            // prefetch next tile
    asm volatile("s_wait_asynccnt 4" ::: "memory");     // current tile done
    __syncthreads();
    compute(cur);
    __syncthreads();
  }
  {                                                     // peeled last tile
    const int cur = (k0 >> 5) & 1;
    asm volatile("s_wait_asynccnt 0" ::: "memory");
    __syncthreads();
    compute(cur);
  }

  // ---- epilogue: VGPR r holds row (r + 8g), col = lane&15 ----
  u16*   C16 = nullptr;
  float* C32 = nullptr;
  if constexpr (OUT_BF16) C16 = (u16*)Cout + (size_t)blockIdx.z * strideC;
  else                    C32 = (float*)Cout + (size_t)blockIdx.z * strideC;

#pragma unroll
  for (int mi = 0; mi < 2; ++mi)
#pragma unroll
    for (int ni = 0; ni < 4; ++ni)
#pragma unroll
      for (int r = 0; r < 8; ++r) {
        int row = bm + wm * 32 + mi * 16 + r + 8 * g;
        int col = bn + wn * 64 + ni * 16 + l15;
        float v = acc[mi][ni][r] * scale;
        if constexpr (ADD_BIAS) v += bias[col];
        if constexpr (OUT_TRANS) {
          // store C^T per batch: [b][col][s]  (V^T for the attn*V GEMM)
          int bidx = row >> 11;            // row / Sdim
          int srow = row & (Sdim - 1);
          C16[((size_t)bidx * Ddim + col) * Sdim + srow] = f2bf(v);
        } else if constexpr (OUT_BF16) {
          C16[(size_t)row * ldc + col] = f2bf(v);
        } else {
          C32[(size_t)row * ldc + col] = v;
        }
      }
}

// ---------------------------------------------------------------------------
__global__ __launch_bounds__(256) void cvt_f32_bf16(const float* __restrict__ in,
                                                    u16* __restrict__ out) {
  size_t i = (size_t)blockIdx.x * 256 + threadIdx.x;
  out[i] = f2bf(in[i]);
}

// W[K][N] fp32 -> Wt[N][K] bf16 (LDS-tiled transpose, 32x32 tiles)
__global__ __launch_bounds__(256) void transpose_f32_bf16(const float* __restrict__ W,
                                                          u16* __restrict__ Wt) {
  __shared__ float t[32][33];
  const int bx = blockIdx.x * 32;         // N
  const int by = blockIdx.y * 32;         // K
  const int tx = threadIdx.x & 31, ty = threadIdx.x >> 5;
#pragma unroll
  for (int j = 0; j < 4; ++j)
    t[ty + 8 * j][tx] = W[(size_t)(by + ty + 8 * j) * Ddim + bx + tx];
  __syncthreads();
#pragma unroll
  for (int j = 0; j < 4; ++j)
    Wt[(size_t)(bx + ty + 8 * j) * Ddim + by + tx] = f2bf(t[tx][ty + 8 * j]);
}

// One block per (b, q) row: softmax over 2048 keys with additive mask input.
__global__ __launch_bounds__(256) void softmax_row(const float* __restrict__ scores,
                                                   const float* __restrict__ mask,
                                                   u16* __restrict__ attn) {
  const int row = blockIdx.x;                 // b*S + q
  const float* srow = scores + (size_t)row * Sdim;
  const float* mrow = mask   + (size_t)row * Sdim;
  u16*         arow = attn   + (size_t)row * Sdim;
  const int t = threadIdx.x;

  float vals[8];
  float mx = -3.0e38f;
#pragma unroll
  for (int i = 0; i < 8; ++i) {
    float v = srow[t + i * 256] + mrow[t + i * 256];
    vals[i] = v;
    mx = fmaxf(mx, v);
  }
  __shared__ float red[256];
  red[t] = mx; __syncthreads();
  for (int s = 128; s > 0; s >>= 1) {
    if (t < s) red[t] = fmaxf(red[t], red[t + s]);
    __syncthreads();
  }
  mx = red[0]; __syncthreads();

  float sum = 0.f;
#pragma unroll
  for (int i = 0; i < 8; ++i) { vals[i] = __expf(vals[i] - mx); sum += vals[i]; }
  red[t] = sum; __syncthreads();
  for (int s = 128; s > 0; s >>= 1) {
    if (t < s) red[t] += red[t + s];
    __syncthreads();
  }
  float inv = 1.0f / red[0];
#pragma unroll
  for (int i = 0; i < 8; ++i) arow[t + i * 256] = f2bf(vals[i] * inv);
}

// ---------------------------------------------------------------------------
extern "C" void kernel_launch(void* const* d_in, const int* in_sizes, int n_in,
                              void* d_out, int out_size, void* d_ws, size_t ws_size,
                              hipStream_t stream) {
  (void)in_sizes; (void)n_in; (void)out_size; (void)ws_size;
  const float* x    = (const float*)d_in[0];
  const float* mask = (const float*)d_in[1];
  const float* Wq   = (const float*)d_in[2];
  const float* bq   = (const float*)d_in[3];
  const float* Wk   = (const float*)d_in[4];
  const float* bk   = (const float*)d_in[5];
  const float* Wv   = (const float*)d_in[6];
  const float* bv   = (const float*)d_in[7];
  float* out = (float*)d_out;

  const size_t NX = (size_t)Bdim * Sdim * Ddim;   // 8,388,608
  const size_t NS = (size_t)Bdim * Sdim * Sdim;   // 16,777,216
  const size_t NW = (size_t)Ddim * Ddim;          // 1,048,576
  char* ws = (char*)d_ws;
  u16*   xbf    = (u16*)ws;                                 // 16 MB
  u16*   qbf    = (u16*)(ws + NX * 2);                      // 16 MB
  u16*   kbf    = (u16*)(ws + NX * 4);                      // 16 MB
  u16*   vtbf   = (u16*)(ws + NX * 6);                      // 16 MB  V^T [B][D][S]
  u16*   wtq    = (u16*)(ws + NX * 8);                      //  2 MB
  u16*   wtk    = (u16*)(ws + NX * 8 + NW * 2);             //  2 MB
  u16*   wtv    = (u16*)(ws + NX * 8 + NW * 4);             //  2 MB
  float* scores = (float*)(ws + NX * 8 + NW * 6);           // 64 MB
  u16*   attn   = (u16*)(ws + NX * 8 + NW * 6 + NS * 4);    // 32 MB  (total 166 MB)

  // 1) x -> bf16
  cvt_f32_bf16<<<(unsigned)(NX / 256), 256, 0, stream>>>(x, xbf);

  // 2) W -> bf16 transposed [N][K] so every GEMM streams B K-contiguously
  dim3 gt(Ddim / 32, Ddim / 32);
  transpose_f32_bf16<<<gt, 256, 0, stream>>>(Wq, wtq);
  transpose_f32_bf16<<<gt, 256, 0, stream>>>(Wk, wtk);
  transpose_f32_bf16<<<gt, 256, 0, stream>>>(Wv, wtv);

  // 3) QKV projections: [8192x1024] = xbf * Wt^T + b
  dim3 gqkv(Ddim / 128, (Bdim * Sdim) / 128, 1);
  gemm_wmma_async<true, true, false><<<gqkv, 256, 0, stream>>>(
      xbf, wtq, bq, qbf, Ddim, Ddim, Ddim, Ddim, 0, 0, 0, 1.0f);
  gemm_wmma_async<true, true, false><<<gqkv, 256, 0, stream>>>(
      xbf, wtk, bk, kbf, Ddim, Ddim, Ddim, Ddim, 0, 0, 0, 1.0f);
  gemm_wmma_async<true, true, true><<<gqkv, 256, 0, stream>>>(   // V stored transposed
      xbf, wtv, bv, vtbf, Ddim, Ddim, Ddim, Ddim, 0, 0, 0, 1.0f);

  // 4) scores = (Q K^T)/sqrt(D); K stored [key][d] == [N][Kd]
  dim3 gsc(Sdim / 128, Sdim / 128, Bdim);
  gemm_wmma_async<false, false, false><<<gsc, 256, 0, stream>>>(
      qbf, kbf, nullptr, scores, Ddim, Ddim, Ddim, Sdim,
      (long long)Sdim * Ddim, (long long)Sdim * Ddim, (long long)Sdim * Sdim,
      0.03125f /* 1/sqrt(1024) */);

  // 5) softmax(scores + mask) -> attn (bf16)
  softmax_row<<<Bdim * Sdim, 256, 0, stream>>>(scores, mask, attn);

  // 6) out = attn * V ; V^T stored [d][key] == [N][Kd]
  dim3 gov(Ddim / 128, Sdim / 128, Bdim);
  gemm_wmma_async<false, false, false><<<gov, 256, 0, stream>>>(
      attn, vtbf, nullptr, out, Sdim, Sdim, Sdim, Ddim,
      (long long)Sdim * Sdim, (long long)Ddim * Sdim, (long long)Sdim * Ddim,
      1.0f);
}
